// ECGLSTM_76656576299321
// MI455X (gfx1250) — matrix operand order repaired and checked by reading, not compiled, not run on hardware
//
#include <hip/hip_runtime.h>
#include <hip/hip_bf16.h>

// ---------------------------------------------------------------------------
// Fused 2-layer bidirectional LSTM (seq_len==1, h0=c0=0) + linear/sigmoid head
// for MI455X (gfx1250, wave32, WMMA f16 16x16x32).
//
// Forget gate and all w_hh are dead (c0==0): each direction is one GEMM over
// the i/g/o gate rows + elementwise. One wave owns 16 rows; both layers and
// the head are fused (h0 lives in a private LDS slice, h1 in registers).
// ---------------------------------------------------------------------------

typedef __attribute__((ext_vector_type(16))) _Float16 v16h;
typedef __attribute__((ext_vector_type(8)))  _Float16 v8h;
typedef __attribute__((ext_vector_type(8)))  float    v8f;

#define NKC   7      // K chunks of 32 (216 or 200 padded to 224)
#define NT    7      // N tiles of 16 per direction (100 padded to 112)
#define KPAD  224
#define WAVES_PER_BLOCK 4
#define LDS_PER_WAVE    8192   // 16*224 f16 h0 tile (7168B) + 1KB reduce scratch

__device__ __forceinline__ float sigm(float x) {
    return 1.0f / (1.0f + __expf(-x));
}

// ---------------------------------------------------------------------------
// Weight pre-pack: w_ih [400, D] f32 (gate rows i:0..99 f:100..199 g:200..299
// o:300..399) -> f16 WMMA B fragments, i/g/o only, K padded to 224.
// Block layout: bi = ((d*NT + t)*3 + g)*NKC + kc, fragment = 32 lanes x 32B.
// ISA 16-bit B 32x16 layout: lane L holds N = L%16, halves h=0..7 hold
// K = kc*32 + 8*(L/16) + h, halves 8..15 hold same + 8 (i.e. +16 for h-8).
// ---------------------------------------------------------------------------
__global__ void ecg_pack_w(const float* __restrict__ wf,
                           const float* __restrict__ wb,
                           int D, v16h* __restrict__ outp) {
    const int lane = threadIdx.x;        // 32 threads/block
    const int bi   = blockIdx.x;         // 2*7*3*7 = 294 blocks
    const int kc   = bi % NKC;
    const int g    = (bi / NKC) % 3;
    const int t    = (bi / (NKC * 3)) % NT;
    const int d    = bi / (NKC * 3 * NT);
    const int gsel[3] = {0, 2, 3};       // i, g, o gate blocks
    const float* __restrict__ w = d ? wb : wf;

    const int nn  = lane & 15;
    const int hi  = lane >> 4;
    const int col = 16 * t + nn;                 // h column (0..111, valid <100)
    const int row = gsel[g] * 100 + col;         // w_ih row
    const int kb  = kc * 32 + hi * 8;

    v16h v;
#pragma unroll
    for (int h = 0; h < 16; ++h) {
        const int k = kb + h + (h >= 8 ? 8 : 0);
        const float f = (col < 100 && k < D) ? w[(size_t)row * D + k] : 0.0f;
        v[h] = (_Float16)f;
    }
    outp[(size_t)bi * 32 + lane] = v;
}

// ---------------------------------------------------------------------------
// Bias / head-weight pre-pack (zero-padded, per-lane layout so the main
// kernel does straight-line non-divergent loads):
//   biasp[(((l*2+d)*NT+t)*3+g)*16 + nn] = b_ih[row] + b_hh[row]  (0 if padded)
//   woutp[(d*NT+t)*16 + nn]            = w_out[d*100+col]        (0 if padded)
// ---------------------------------------------------------------------------
__global__ void ecg_pack_misc(const float* __restrict__ bi0f, const float* __restrict__ bh0f,
                              const float* __restrict__ bi0b, const float* __restrict__ bh0b,
                              const float* __restrict__ bi1f, const float* __restrict__ bh1f,
                              const float* __restrict__ bi1b, const float* __restrict__ bh1b,
                              const float* __restrict__ wout,
                              float* __restrict__ biasp, float* __restrict__ woutp) {
    const int gsel[3] = {0, 2, 3};
    const int tid  = threadIdx.x + blockIdx.x * blockDim.x;
    const int step = blockDim.x * gridDim.x;
    for (int i = tid; i < 2 * 2 * NT * 3 * 16; i += step) {   // 1344
        const int nn = i & 15;
        const int g  = (i >> 4) % 3;
        const int t  = (i / 48) % NT;
        const int d  = (i / (48 * NT)) & 1;
        const int l  = i / (48 * NT * 2);
        const int col = 16 * t + nn;
        float v = 0.0f;
        if (col < 100) {
            const int row = gsel[g] * 100 + col;
            const float* bi = l ? (d ? bi1b : bi1f) : (d ? bi0b : bi0f);
            const float* bh = l ? (d ? bh1b : bh1f) : (d ? bh0b : bh0f);
            v = bi[row] + bh[row];
        }
        biasp[i] = v;
    }
    for (int i = tid; i < 2 * NT * 16; i += step) {           // 224
        const int nn = i & 15;
        const int t  = (i >> 4) % NT;
        const int d  = i / (16 * NT);
        const int col = 16 * t + nn;
        woutp[i] = (col < 100) ? wout[d * 100 + col] : 0.0f;
    }
}

// ---------------------------------------------------------------------------
// One LSTM layer over a 16-row tile held as A fragments in registers.
// Bias is folded into the WMMA C initializer (C layout: lane == column).
// L==0: stores h tile (f16) into LDS h0 buffer [16][KPAD], cols d*100+col.
// L==1: folds head dot-product into headacc[8].
// ---------------------------------------------------------------------------
template <int L>
__device__ __forceinline__ void run_layer(
    const v16h* __restrict__ Wp, const v16h a[NKC],
    const float* __restrict__ bslab,     // this layer's 42*16 packed biases
    int lane, _Float16* __restrict__ h0out,
    const float* __restrict__ woutp, float* headacc) {
    const int nn = lane & 15;
    const int hi = lane >> 4;

    for (int d = 0; d < 2; ++d) {
        for (int t = 0; t < NT; ++t) {
            const int  col16 = 16 * t + nn;
            const bool valid = (col16 < 100);
            const float* __restrict__ bptr =
                bslab + ((d * NT + t) * 3) * 16 + nn;
            v8f acc[3];
#pragma unroll
            for (int g = 0; g < 3; ++g) {
                const float bv = bptr[g * 16];
                v8f c = {bv, bv, bv, bv, bv, bv, bv, bv};
                const size_t bbase =
                    ((size_t)((d * NT + t) * 3 + g) * NKC) * 32 + lane;
                // Stage all 7 B fragments first: one load clause, then the
                // WMMA chain drains with staggered s_wait_loadcnt.
                v16h b[NKC];
#pragma unroll
                for (int kc = 0; kc < NKC; ++kc)
                    b[kc] = Wp[bbase + (size_t)kc * 32];
#pragma unroll
                for (int kc = 0; kc < NKC; ++kc)
                    c = __builtin_amdgcn_wmma_f32_16x16x32_f16(
                        false, a[kc], false, b[kc], (short)0, c, false, false);
                acc[g] = c;
            }
            float wo = 0.0f;
            if (L == 1) wo = woutp[(d * NT + t) * 16 + nn];   // zero-padded
#pragma unroll
            for (int r = 0; r < 8; ++r) {
                const float cc = sigm(acc[0][r]) * tanhf(acc[1][r]);
                const float h  = sigm(acc[2][r]) * tanhf(cc);
                if (L == 0) {
                    if (valid)
                        h0out[(r + hi * 8) * KPAD + d * 100 + col16] =
                            (_Float16)h;
                } else {
                    headacc[r] += h * wo;
                }
            }
        }
    }
}

// ---------------------------------------------------------------------------
// Fused main kernel: 1 wave per 16 input rows, 4 waves per block,
// private 8KB LDS slice per wave (no barriers needed).
// ---------------------------------------------------------------------------
__global__ void __launch_bounds__(WAVES_PER_BLOCK * 32)
ecg_lstm_fused(const float* __restrict__ x,
               const v16h* __restrict__ W0p, const v16h* __restrict__ W1p,
               const float* __restrict__ biasp, const float* __restrict__ woutp,
               const float* __restrict__ bout,
               float* __restrict__ out, int N) {
    extern __shared__ char smem[];
    const int lane = threadIdx.x & 31;
    const int wid  = threadIdx.x >> 5;
    const long m0  = ((long)blockIdx.x * WAVES_PER_BLOCK + wid) * 16;
    if (m0 >= N) return;

    _Float16* __restrict__ h0  = (_Float16*)(smem + wid * LDS_PER_WAVE);
    float*    __restrict__ red = (float*)(smem + wid * LDS_PER_WAVE + 7168);

    const int nn = lane & 15;
    const int hi = lane >> 4;

    // ---- A0 fragments from x (f32 -> f16), ISA 16-bit A 16x32 layout:
    // lane L (M = L%16): halves 0..7 -> K = kc*32 + 8*(L/16) + j,
    //                    halves 8..15 -> same + 16.
    v16h a0[NKC];
    const float* __restrict__ xr = x + (size_t)(m0 + nn) * 216;
#pragma unroll
    for (int kc = 0; kc < NKC; ++kc) {
        const int kb = kc * 32 + hi * 8;
#pragma unroll
        for (int j = 0; j < 8; ++j) {
            const int k0 = kb + j;
            const int k1 = kb + 16 + j;
            a0[kc][j]     = (k0 < 216) ? (_Float16)xr[k0] : (_Float16)0.0f;
            a0[kc][8 + j] = (k1 < 216) ? (_Float16)xr[k1] : (_Float16)0.0f;
        }
    }

    // Zero the K padding columns [200, 224) of the h0 tile.
    for (int idx = lane; idx < 16 * 24; idx += 32) {
        const int r = idx / 24, c = 200 + idx % 24;
        h0[r * KPAD + c] = (_Float16)0.0f;
    }

    // ---- Layer 0 (both directions) -> h0 in LDS
    run_layer<0>(W0p, a0, biasp, lane, h0, nullptr, nullptr);

    // ---- A1 fragments from LDS h0 (aligned 16B f16 chunks)
    v16h a1[NKC];
    const _Float16* __restrict__ myrow = h0 + nn * KPAD;
#pragma unroll
    for (int kc = 0; kc < NKC; ++kc) {
        const int kb = kc * 32 + hi * 8;
        const v8h lo  = *(const v8h*)(myrow + kb);
        const v8h hi8 = *(const v8h*)(myrow + kb + 16);
#pragma unroll
        for (int j = 0; j < 8; ++j) {
            a1[kc][j]     = lo[j];
            a1[kc][8 + j] = hi8[j];
        }
    }

    // ---- Layer 1 + fused head accumulation
    float headacc[8];
#pragma unroll
    for (int r = 0; r < 8; ++r) headacc[r] = 0.0f;
    run_layer<1>(W1p, a1, biasp + 2 * NT * 3 * 16, lane, nullptr, woutp,
                 headacc);

    // ---- Head reduction: acc[r] on lanes 0..15 -> row r, lanes 16..31 -> 8+r
#pragma unroll
    for (int r = 0; r < 8; ++r) red[lane * 8 + r] = headacc[r];
    if (lane < 16) {
        const int r = lane & 7, half = lane >> 3;
        float s = 0.0f;
#pragma unroll
        for (int j = 0; j < 16; ++j) s += red[(half * 16 + j) * 8 + r];
        out[m0 + lane] = sigm(s + bout[0]);
    }
}

// ---------------------------------------------------------------------------
// Launch
// ---------------------------------------------------------------------------
extern "C" void kernel_launch(void* const* d_in, const int* in_sizes, int n_in,
                              void* d_out, int out_size, void* d_ws, size_t ws_size,
                              hipStream_t stream) {
    const float* x      = (const float*)d_in[0];
    const float* w_ih0f = (const float*)d_in[1];
    const float* b_ih0f = (const float*)d_in[3];
    const float* b_hh0f = (const float*)d_in[4];
    const float* w_ih0b = (const float*)d_in[5];
    const float* b_ih0b = (const float*)d_in[7];
    const float* b_hh0b = (const float*)d_in[8];
    const float* w_ih1f = (const float*)d_in[9];
    const float* b_ih1f = (const float*)d_in[11];
    const float* b_hh1f = (const float*)d_in[12];
    const float* w_ih1b = (const float*)d_in[13];
    const float* b_ih1b = (const float*)d_in[15];
    const float* b_hh1b = (const float*)d_in[16];
    const float* w_out  = (const float*)d_in[17];
    const float* b_out  = (const float*)d_in[18];

    const int N = in_sizes[0] / 216;

    const size_t wpack_bytes = (size_t)2 * NT * 3 * NKC * 1024;  // 301056 B
    v16h*  W0p   = (v16h*)d_ws;
    v16h*  W1p   = (v16h*)((char*)d_ws + wpack_bytes);
    float* biasp = (float*)((char*)d_ws + 2 * wpack_bytes);
    float* woutp = biasp + 2 * 2 * NT * 3 * 16;                  // +1344 floats

    ecg_pack_w<<<2 * NT * 3 * NKC, 32, 0, stream>>>(w_ih0f, w_ih0b, 216, W0p);
    ecg_pack_w<<<2 * NT * 3 * NKC, 32, 0, stream>>>(w_ih1f, w_ih1b, 200, W1p);
    ecg_pack_misc<<<2, 256, 0, stream>>>(b_ih0f, b_hh0f, b_ih0b, b_hh0b,
                                         b_ih1f, b_hh1f, b_ih1b, b_hh1b,
                                         w_out, biasp, woutp);

    const int waves  = (N + 15) / 16;
    const int blocks = (waves + WAVES_PER_BLOCK - 1) / WAVES_PER_BLOCK;
    ecg_lstm_fused<<<blocks, WAVES_PER_BLOCK * 32,
                     WAVES_PER_BLOCK * LDS_PER_WAVE, stream>>>(
        x, W0p, W1p, biasp, woutp, b_out, (float*)d_out, N);
}
